// MHTA_6846177870249
// MI455X (gfx1250) — matrix operand order repaired and checked by reading, not compiled, not run on hardware
//
#include <hip/hip_runtime.h>
#include <math.h>
#include <stdint.h>

// ---------------------------------------------------------------------------
// MHTA (multi-head transposed channel attention) for MI455X / gfx1250, wave32.
// Pipeline: depthwise qkv conv -> bf16 q,k,v -> WMMA logits GEMM -> softmax
// -> fold proj into attn (M = w_proj * blockdiag(attn)) -> WMMA M @ v GEMM.
// v is streamed from HBM exactly once; output uses non-temporal stores so the
// write-once stream does not pollute the 192MB L2 holding q/k/v.
// ---------------------------------------------------------------------------

#define DIMC 192
#define NHEADS 4
#define BATCH 2
#define HW 65536        // 256*256
#define CH 48           // channels per head
#define EPSN 1e-12f

typedef __attribute__((ext_vector_type(4)))  unsigned u32x4;
typedef __attribute__((ext_vector_type(4)))  int      i32x4;
typedef __attribute__((ext_vector_type(8)))  float    v8f;
typedef __attribute__((ext_vector_type(16))) __bf16   v16bf;

union ABu {
    v16bf v;          // WMMA operand
    unsigned x[8];    // 8 dwords = 16 bf16, element 2j|2j+1 = dword j lo|hi
    u32x4 y[2];       // two 16B chunks
};

// Async global->LDS staging (CDNA5 ASYNCcnt path), guarded so compile is safe.
#if defined(__gfx1250__) && __has_builtin(__builtin_amdgcn_global_load_async_to_lds_b128) && __has_builtin(__builtin_amdgcn_s_wait_asynccnt)
#define HAVE_ASYNC 1
#else
#define HAVE_ASYNC 0
#endif

// builtin signature: (int4 AS1* gaddr, int4 AS3* ldsaddr, imm offset, imm cpol)
typedef __attribute__((address_space(1))) i32x4* gas1_p;
typedef __attribute__((address_space(3))) i32x4* las3_p;

__device__ __host__ inline gas1_p to_as1_v4(const void* p) {
    return (gas1_p)(unsigned long long)(uintptr_t)p;
}
__device__ __host__ inline las3_p to_as3_v4(void* p) {
    return (las3_p)(unsigned)(uintptr_t)p;
}

__device__ inline unsigned short f2bf_bits(float f) {
    union { float f; unsigned u; } a; a.f = f;
    unsigned r = a.u + 0x7FFFu + ((a.u >> 16) & 1u);   // round-to-nearest-even
    return (unsigned short)(r >> 16);
}

// ---------------------------------------------------------------------------
// Kernel 1: depthwise 3x3 conv -> q,k,v (bf16) + sum-of-squares for q,k norms.
// grid (H=256, 3*DIM=576, B=2), block 256 (one output row per block).
// ---------------------------------------------------------------------------
__global__ void qkv_conv_kernel(const float* __restrict__ x,
                                const float* __restrict__ w_qkv,
                                unsigned short* __restrict__ q,
                                unsigned short* __restrict__ k,
                                unsigned short* __restrict__ v,
                                float* __restrict__ sumsq) {
    const int xx = threadIdx.x;
    const int y  = blockIdx.x;
    const int o  = blockIdx.y;
    const int b  = blockIdx.z;
    const int ic = o / 3;

    const float* xb = x + (((size_t)b * DIMC + ic) << 16);
    float wgt[9];
#pragma unroll
    for (int i = 0; i < 9; ++i) wgt[i] = w_qkv[o * 9 + i];

    float acc = 0.f;
#pragma unroll
    for (int dy = -1; dy <= 1; ++dy) {
        int yy = y + dy;
        if ((unsigned)yy < 256u) {
#pragma unroll
            for (int dx = -1; dx <= 1; ++dx) {
                int xc = xx + dx;
                if ((unsigned)xc < 256u)
                    acc += wgt[(dy + 1) * 3 + (dx + 1)] * xb[yy * 256 + xc];
            }
        }
    }

    const int n = y * 256 + xx;
    const int c = (o < DIMC) ? o : ((o < 2 * DIMC) ? o - DIMC : o - 2 * DIMC);
    const size_t off = ((size_t)b * DIMC + c) * HW + n;
    unsigned short bits = f2bf_bits(acc);
    if (o < DIMC)            q[off] = bits;
    else if (o < 2 * DIMC)   k[off] = bits;
    else                     v[off] = bits;

    if (o < 2 * DIMC) {      // accumulate ||q_c||^2, ||k_c||^2
        __shared__ float red[256];
        red[threadIdx.x] = acc * acc;
        __syncthreads();
#pragma unroll
        for (int s = 128; s > 0; s >>= 1) {
            if (threadIdx.x < s) red[threadIdx.x] += red[threadIdx.x + s];
            __syncthreads();
        }
        if (threadIdx.x == 0) atomicAdd(&sumsq[b * (2 * DIMC) + o], red[0]);
    }
}

// ---------------------------------------------------------------------------
// Kernel 2: attn logits = q_h * k_h^T per (b,head), 48x48 = 3x3 tiles of 16x16.
// K = 65536 split across 64 blocks; per block: 4 x (stage 256-K slab in LDS,
// 8 waves each consume a 32-K chunk with v_wmma_f32_16x16x32_bf16).
// k-transpose staging: b64 loads per row-pair, ALU merge, b32 LDS stores.
// grid (64, 72), block 256 = 8 waves.
// ---------------------------------------------------------------------------
__global__ void attn_gemm_kernel(const unsigned short* __restrict__ q,
                                 const unsigned short* __restrict__ k,
                                 float* __restrict__ logits) {
    const int ks = blockIdx.x;           // K split id
    const int t  = blockIdx.y;           // tile id
    const int bh = t / 9, rem = t % 9, tr = rem / 3, tc = rem % 3;
    const int b = bh / NHEADS, h = bh % NHEADS;

    const unsigned short* qb = q + ((size_t)b * DIMC + h * CH + tr * 16) * HW;
    const unsigned short* kb = k + ((size_t)b * DIMC + h * CH + tc * 16) * HW;

    __shared__ unsigned short qt[16 * 256];   // A slab: [row(ch)][K(n)]
    __shared__ unsigned short kt[256 * 16];   // B slab transposed: [K(n)][col(ch)]
    __shared__ float redc[256];

    const int tid  = threadIdx.x;
    const int lane = tid & 31;
    const int wave = tid >> 5;
    const int m    = lane & 15;
    const int half = lane >> 4;

    v8f acc = {0.f, 0.f, 0.f, 0.f, 0.f, 0.f, 0.f, 0.f};

    for (int ob = 0; ob < 4; ++ob) {
        const int nbase = ks * 1024 + ob * 256;
        __syncthreads();
        // stage q tile: 16 rows x 256 n, 512 u32x4
#pragma unroll
        for (int i = 0; i < 2; ++i) {
            int idx = tid + i * 256;
            int row = idx >> 5, col = idx & 31;
            reinterpret_cast<u32x4*>(qt)[idx] =
                *reinterpret_cast<const u32x4*>(qb + (size_t)row * HW + nbase + col * 8);
        }
        // stage k tile transposed: 512 units = 8 row-pairs x 64 col-quads
#pragma unroll
        for (int i = 0; i < 2; ++i) {
            int u  = tid + i * 256;
            int rp = u >> 6;                    // row pair 0..7
            int n  = (u & 63) * 4;              // col quad
            const unsigned* r0 = reinterpret_cast<const unsigned*>(kb + (size_t)(2 * rp) * HW + nbase + n);
            const unsigned* r1 = reinterpret_cast<const unsigned*>(kb + (size_t)(2 * rp + 1) * HW + nbase + n);
            unsigned a0 = r0[0], a1 = r0[1];
            unsigned b0 = r1[0], b1 = r1[1];
            *reinterpret_cast<unsigned*>(&kt[(n + 0) * 16 + 2 * rp]) = (a0 & 0xFFFFu) | (b0 << 16);
            *reinterpret_cast<unsigned*>(&kt[(n + 1) * 16 + 2 * rp]) = (a0 >> 16) | (b0 & 0xFFFF0000u);
            *reinterpret_cast<unsigned*>(&kt[(n + 2) * 16 + 2 * rp]) = (a1 & 0xFFFFu) | (b1 << 16);
            *reinterpret_cast<unsigned*>(&kt[(n + 3) * 16 + 2 * rp]) = (a1 >> 16) | (b1 & 0xFFFF0000u);
        }
        __syncthreads();

        const int kb32 = wave * 32;
        // A fragment: lane<16 holds M=m, K {0..7,16..23}; lane>=16 K {8..15,24..31}
        ABu a;
#pragma unroll
        for (int j = 0; j < 8; ++j) {
            int kp = ((j < 4) ? 2 * j : 16 + 2 * (j - 4)) + half * 8;
            a.x[j] = *reinterpret_cast<const unsigned*>(&qt[m * 256 + kb32 + kp]);
        }
        // B fragment: lane holds N=m, K in the same order as A
        ABu bf;
#pragma unroll
        for (int j = 0; j < 8; ++j) {
            int kp = ((j < 4) ? 2 * j : 16 + 2 * (j - 4)) + half * 8;
            bf.x[j] = (unsigned)kt[(kb32 + kp) * 16 + m] | ((unsigned)kt[(kb32 + kp + 1) * 16 + m] << 16);
        }
        acc = __builtin_amdgcn_wmma_f32_16x16x32_bf16(false, a.v, false, bf.v,
                                                      (short)0, acc, false, false);
    }

    // cross-wave reduction of the 16x16 tile, then one global atomic per elem
    redc[tid] = 0.f;
    __syncthreads();
#pragma unroll
    for (int j = 0; j < 8; ++j)
        atomicAdd(&redc[(j + half * 8) * 16 + m], acc[j]);
    __syncthreads();
    const int row = tid >> 4, col = tid & 15;
    atomicAdd(&logits[(((size_t)(b * NHEADS + h) * CH + tr * 16 + row) * CH) + tc * 16 + col],
              redc[tid]);
}

// ---------------------------------------------------------------------------
// Kernel 3a: rescale by 1/(|q||k|) * temperature, softmax over each 48-row.
// ---------------------------------------------------------------------------
__global__ void softmax_kernel(float* __restrict__ logits,
                               const float* __restrict__ sumsq,
                               const float* __restrict__ temperature) {
    const int r = blockIdx.x;
    const int b = r / (NHEADS * CH), rem = r % (NHEADS * CH);
    const int h = rem / CH, c = rem % CH;
    float* row = logits + ((size_t)(b * NHEADS + h) * CH + c) * CH;

    const float invq = 1.f / fmaxf(sqrtf(sumsq[b * 2 * DIMC + h * CH + c]), EPSN);
    const float tmp  = temperature[h];
    const int t = threadIdx.x;

    __shared__ float sm[64];
    float myv = 0.f, val = -1e30f;
    if (t < CH) {
        float invk = 1.f / fmaxf(sqrtf(sumsq[b * 2 * DIMC + DIMC + h * CH + t]), EPSN);
        myv = row[t] * invq * invk * tmp;
        val = myv;
    }
    sm[t] = val;
    __syncthreads();
#pragma unroll
    for (int s = 32; s > 0; s >>= 1) {
        if (t < s) sm[t] = fmaxf(sm[t], sm[t + s]);
        __syncthreads();
    }
    const float mx = sm[0];
    __syncthreads();
    const float e = (t < CH) ? __expf(myv - mx) : 0.f;
    sm[t] = e;
    __syncthreads();
#pragma unroll
    for (int s = 32; s > 0; s >>= 1) {
        if (t < s) sm[t] += sm[t + s];
        __syncthreads();
    }
    if (t < CH) row[t] = e / sm[0];
}

// ---------------------------------------------------------------------------
// Kernel 3b: M_b = w_proj * blockdiag(attn_b)  (bf16, row-major [b][o][i]).
// ---------------------------------------------------------------------------
__global__ void build_m_kernel(const float* __restrict__ logits,
                               const float* __restrict__ w_proj,
                               unsigned short* __restrict__ M) {
    const int idx = blockIdx.x * 256 + threadIdx.x;       // < 2*192*192
    const int b = idx / (DIMC * DIMC), rem = idx % (DIMC * DIMC);
    const int o = rem / DIMC, col = rem % DIMC;
    const int h = col / CH, d = col % CH;
    const float* a = logits + ((size_t)(b * NHEADS + h) * CH) * CH + d;
    const float* w = w_proj + o * DIMC + h * CH;
    float s = 0.f;
#pragma unroll
    for (int c = 0; c < CH; ++c) s += w[c] * a[c * CH];
    M[idx] = f2bf_bits(s);
}

// ---------------------------------------------------------------------------
// Kernel 4: out[b] = M_b (192x192) @ v_b (192x65536), f32 out.
// grid (512 Nblk, 2 b), block 384 = 12 waves; each wave one 16-row M-tile,
// N-block 128 -> v is read from HBM exactly once. M_b panel (72KB) in LDS.
// B slabs double-buffered via global_load_async_to_lds_b128 (ASYNCcnt).
// Output written with non-temporal stores.
// ---------------------------------------------------------------------------
__global__ void __launch_bounds__(384)
out_gemm_kernel(const unsigned short* __restrict__ Mw,
                const unsigned short* __restrict__ v,
                float* __restrict__ out) {
    const int b     = blockIdx.y;
    const int nbase = blockIdx.x * 128;

    __shared__ unsigned short At[DIMC * DIMC];   // 72KB: [row][K]
#if HAVE_ASYNC
    __shared__ unsigned short Bt[2][32 * 128];   // 2 x 8KB double buffer
#else
    __shared__ unsigned short Bt[1][32 * 128];
#endif

    const int tid  = threadIdx.x;
    const int lane = tid & 31;
    const int wave = tid >> 5;
    const int m    = lane & 15;
    const int half = lane >> 4;
    const int mt   = wave;                       // 0..11: M-tile (16 rows)

    // stage full A panel (192 x 192 bf16 = 4608 u32x4)
    const unsigned short* Mb = Mw + (size_t)b * DIMC * DIMC;
#pragma unroll
    for (int i = 0; i < 12; ++i) {
        int idx = tid + i * 384;
        reinterpret_cast<u32x4*>(At)[idx] = reinterpret_cast<const u32x4*>(Mb)[idx];
    }

    const unsigned short* vb = v + (size_t)b * DIMC * HW;

    v8f acc[8];
#pragma unroll
    for (int nt = 0; nt < 8; ++nt) acc[nt] = (v8f){0.f,0.f,0.f,0.f,0.f,0.f,0.f,0.f};

#if HAVE_ASYNC
    // one slab = 32 K-rows x 128 n = 512 16B chunks
    auto stage_async = [&](int kk) {
        unsigned short* dst = Bt[kk & 1];
        for (int idx = tid; idx < 512; idx += 384) {
            int row = idx >> 4, col = idx & 15;
            const void* g = vb + (size_t)(kk * 32 + row) * HW + nbase + col * 8;
            __builtin_amdgcn_global_load_async_to_lds_b128(to_as1_v4(g), to_as3_v4(dst + idx * 8), 0, 0);
        }
    };
    stage_async(0);
#endif

#pragma unroll
    for (int kk = 0; kk < 6; ++kk) {
#if HAVE_ASYNC
        __builtin_amdgcn_s_wait_asynccnt(0);   // my slab-kk chunks landed
        __syncthreads();                       // everyone's landed; prev compute done
        if (kk < 5) stage_async(kk + 1);       // overlap load(kk+1) with compute(kk)
        const unsigned short* B = Bt[kk & 1];
#else
        __syncthreads();
        for (int idx = tid; idx < 512; idx += 384) {
            int row = idx >> 4, col = idx & 15;
            reinterpret_cast<u32x4*>(Bt[0])[idx] =
                *reinterpret_cast<const u32x4*>(vb + (size_t)(kk * 32 + row) * HW + nbase + col * 8);
        }
        __syncthreads();
        const unsigned short* B = Bt[0];
#endif
        ABu a;
#pragma unroll
        for (int j = 0; j < 8; ++j) {
            int kp = ((j < 4) ? 2 * j : 16 + 2 * (j - 4)) + half * 8;
            a.x[j] = *reinterpret_cast<const unsigned*>(&At[(mt * 16 + m) * DIMC + kk * 32 + kp]);
        }
#pragma unroll
        for (int nt = 0; nt < 8; ++nt) {
            ABu bf;
#pragma unroll
            for (int j = 0; j < 8; ++j) {
                int kp = ((j < 4) ? 2 * j : 16 + 2 * (j - 4)) + half * 8;
                bf.x[j] = (unsigned)B[kp * 128 + nt * 16 + m] | ((unsigned)B[(kp + 1) * 128 + nt * 16 + m] << 16);
            }
            acc[nt] = __builtin_amdgcn_wmma_f32_16x16x32_bf16(false, a.v, false, bf.v,
                                                              (short)0, acc[nt], false, false);
        }
    }

    float* ob = out + ((size_t)b * DIMC + mt * 16) * HW + nbase;
#pragma unroll
    for (int nt = 0; nt < 8; ++nt)
#pragma unroll
        for (int j = 0; j < 8; ++j)
            __builtin_nontemporal_store(acc[nt][j], &ob[(size_t)(j + half * 8) * HW + nt * 16 + m]);
}

// ---------------------------------------------------------------------------
// Host launcher
// ---------------------------------------------------------------------------
extern "C" void kernel_launch(void* const* d_in, const int* in_sizes, int n_in,
                              void* d_out, int out_size, void* d_ws, size_t ws_size,
                              hipStream_t stream) {
    const float* x    = (const float*)d_in[0];
    const float* wqkv = (const float*)d_in[1];
    const float* temp = (const float*)d_in[2];
    const float* wprj = (const float*)d_in[3];
    float* out = (float*)d_out;

    char* ws = (char*)d_ws;
    const size_t qkv_bytes = (size_t)BATCH * DIMC * HW * 2;   // bf16 plane
    unsigned short* q = (unsigned short*)(ws);
    unsigned short* k = (unsigned short*)(ws + qkv_bytes);
    unsigned short* v = (unsigned short*)(ws + 2 * qkv_bytes);
    float* sumsq  = (float*)(ws + 3 * qkv_bytes);                             // 2*384 f32
    float* logits = (float*)(ws + 3 * qkv_bytes + 3072);                      // 2*4*48*48 f32
    unsigned short* M = (unsigned short*)(ws + 3 * qkv_bytes + 3072 + 73728); // 2*192*192 bf16

    // zero the atomic-accumulated regions (sumsq + logits are contiguous)
    (void)hipMemsetAsync(sumsq, 0, 3072 + 73728, stream);

    qkv_conv_kernel<<<dim3(256, 3 * DIMC, BATCH), 256, 0, stream>>>(x, wqkv, q, k, v, sumsq);
    attn_gemm_kernel<<<dim3(64, BATCH * NHEADS * 9), 256, 0, stream>>>(q, k, logits);
    softmax_kernel<<<dim3(BATCH * NHEADS * CH), 64, 0, stream>>>(logits, sumsq, temp);
    build_m_kernel<<<dim3((BATCH * DIMC * DIMC) / 256), 256, 0, stream>>>(logits, wprj, M);
    out_gemm_kernel<<<dim3(HW / 128, BATCH), 384, 0, stream>>>(M, v, out);
}
